// Head_79027398246681
// MI455X (gfx1250) — matrix operand order repaired and checked by reading, not compiled, not run on hardware
//
#include <hip/hip_runtime.h>
#include <hip/hip_bf16.h>

// Fused causal single-head attention for MI455X (gfx1250, wave32, WMMA).
// B=4096, T=96, C=256, H=64. One workgroup (6 wave32) per batch element.
// All GEMMs run on v_wmma_f32_16x16x32_bf16 with hi/lo bf16 split inputs
// (3 WMMAs per logical fp32 MAC tile -> ~fp32 accuracy on the fast path).

typedef __attribute__((ext_vector_type(16))) __bf16        v16bf;
typedef __attribute__((ext_vector_type(8)))  float         v8f;
typedef __attribute__((ext_vector_type(4)))  unsigned int  u32x4;

union FragU { v16bf v; u32x4 q[2]; };

__device__ __forceinline__ unsigned short f2bf(float f) {
  unsigned u = __float_as_uint(f);
  unsigned r = u + 0x7FFFu + ((u >> 16) & 1u);   // round-to-nearest-even
  return (unsigned short)(r >> 16);
}
__device__ __forceinline__ float bf2f(unsigned short h) {
  return __uint_as_float(((unsigned)h) << 16);
}

// ---- geometry ----
static constexpr int T = 96, C = 256, H = 64;
static constexpr int NWAVE = 6, NTHR = NWAVE * 32;

// ---- LDS layout (bytes). Phase 2 aliases phase 1 (barriers separate them).
static constexpr int PX   = C + 8;                 // 264 bf16 pitch (x rows, Wt rows)
static constexpr int SZ_X = T * PX * 2;            // 50688
static constexpr int OFF_XH = 0;
static constexpr int OFF_XL = OFF_XH + SZ_X;
static constexpr int SZ_W = 3 * H * PX * 2;        // 101376 (3 weights, transposed HxC)
static constexpr int OFF_WH = OFF_XL + SZ_X;       // 101376
static constexpr int OFF_WL = OFF_WH + SZ_W;       // 202752
static constexpr int SMEM_BYTES = OFF_WL + SZ_W;   // 304128 (< 320KB WGP LDS)

static constexpr int PQK  = H + 8;                 // 72  (Q,K rows [T][H])
static constexpr int PVT  = T + 8;                 // 104 (V^T rows [H][T])
static constexpr int PW32 = T + 4;                 // 100 (f32 wei pitch)
static constexpr int PWB  = T + 8;                 // 104 (bf16 wei pitch)
static constexpr int SZ_QK = T * PQK * 2;          // 13824
static constexpr int OFF_QH = 0;
static constexpr int OFF_QL  = OFF_QH + SZ_QK;
static constexpr int OFF_KH  = OFF_QL + SZ_QK;
static constexpr int OFF_KL  = OFF_KH + SZ_QK;
static constexpr int SZ_VT   = H * PVT * 2;        // 13312
static constexpr int OFF_VTH = OFF_KL + SZ_QK;
static constexpr int OFF_VTL = OFF_VTH + SZ_VT;
static constexpr int OFF_WEI = OFF_VTL + SZ_VT;    // f32 logits
static constexpr int SZ_WEI  = T * PW32 * 4;       // 38400
static constexpr int OFF_WBH = OFF_WEI + SZ_WEI;
static constexpr int SZ_WB   = T * PWB * 2;        // 19968
static constexpr int OFF_WBL = OFF_WBH + SZ_WB;    // end 160256 <= SMEM_BYTES

// A fragment (16x32 bf16, ISA 16-bit A layout): lane m=L%16.., per-lane K
// groups of 8: elems 0..7 <- K=k0+8*half.. , elems 8..15 <- K=k0+16+8*half..
__device__ __forceinline__ v16bf loadA(const unsigned short* base, int row,
                                       int pitch, int k0, int half) {
  FragU f;
  const unsigned short* p = base + row * pitch + k0 + 8 * half;
  f.q[0] = *(const u32x4*)p;
  f.q[1] = *(const u32x4*)(p + 16);
  return f.v;
}
// B fragment (32x16 bf16): lane n, elems 0..7 <- K = k0 + 8*half + 0..7,
// elems 8..15 <- K = k0 + 16 + 8*half + 0..7 (same K order as A). Source arrays
// are stored transposed (rows = n, contiguous in K).
__device__ __forceinline__ v16bf loadB(const unsigned short* baseT, int col,
                                       int pitch, int k0, int half) {
  FragU f;
  const unsigned short* p = baseT + col * pitch + k0 + 8 * half;
  f.q[0] = *(const u32x4*)p;
  f.q[1] = *(const u32x4*)(p + 16);
  return f.v;
}

// (Ah+Al)(Bh+Bl) ~= AhBh + AhBl + AlBh   (AlBl ~ 2^-18, dropped)
__device__ __forceinline__ v8f wmma3(v16bf ah, v16bf al, v16bf bh, v16bf bl, v8f c) {
  c = __builtin_amdgcn_wmma_f32_16x16x32_bf16(false, ah, false, bh, (short)0, c, false, false);
  c = __builtin_amdgcn_wmma_f32_16x16x32_bf16(false, ah, false, bl, (short)0, c, false, false);
  c = __builtin_amdgcn_wmma_f32_16x16x32_bf16(false, al, false, bh, (short)0, c, false, false);
  return c;
}

__global__ void __launch_bounds__(NTHR, 1)
attn_head_wmma(const float* __restrict__ X,  const float* __restrict__ Wk,
               const float* __restrict__ Wq, const float* __restrict__ Wv,
               float* __restrict__ Out)
{
  extern __shared__ char smem[];
  unsigned short* sXh  = (unsigned short*)(smem + OFF_XH);
  unsigned short* sXl  = (unsigned short*)(smem + OFF_XL);
  unsigned short* sWh  = (unsigned short*)(smem + OFF_WH);
  unsigned short* sWl  = (unsigned short*)(smem + OFF_WL);
  unsigned short* sQh  = (unsigned short*)(smem + OFF_QH);
  unsigned short* sQl  = (unsigned short*)(smem + OFF_QL);
  unsigned short* sKh  = (unsigned short*)(smem + OFF_KH);
  unsigned short* sKl  = (unsigned short*)(smem + OFF_KL);
  unsigned short* sVth = (unsigned short*)(smem + OFF_VTH);
  unsigned short* sVtl = (unsigned short*)(smem + OFF_VTL);
  float*          sWei = (float*)        (smem + OFF_WEI);
  unsigned short* sWbh = (unsigned short*)(smem + OFF_WBH);
  unsigned short* sWbl = (unsigned short*)(smem + OFF_WBL);

  const int b    = blockIdx.x;
  const int tid  = threadIdx.x;
  const int lane = tid & 31;
  const int wv   = tid >> 5;        // wave id 0..5 == M-tile id
  const int half = lane >> 4;
  const int ml   = lane & 15;

  // ---- stage x[b] as hi/lo bf16 (coalesced b128 reads) ----
  const float* xb = X + (size_t)b * T * C;
  for (int i = tid; i < T * C / 4; i += NTHR) {
    int row = i >> 6;               // / (C/4)
    int c4  = (i & 63) << 2;
    float4 f = ((const float4*)xb)[i];
    float vs[4] = {f.x, f.y, f.z, f.w};
    unsigned int hh[2], ll[2];
    unsigned short hs[4], ls[4];
    #pragma unroll
    for (int j = 0; j < 4; ++j) {
      hs[j] = f2bf(vs[j]);
      ls[j] = f2bf(vs[j] - bf2f(hs[j]));
    }
    hh[0] = (unsigned)hs[0] | ((unsigned)hs[1] << 16);
    hh[1] = (unsigned)hs[2] | ((unsigned)hs[3] << 16);
    ll[0] = (unsigned)ls[0] | ((unsigned)ls[1] << 16);
    ll[1] = (unsigned)ls[2] | ((unsigned)ls[3] << 16);
    *(uint2*)(sXh + row * PX + c4) = make_uint2(hh[0], hh[1]);
    *(uint2*)(sXl + row * PX + c4) = make_uint2(ll[0], ll[1]);
  }
  // ---- stage W (global [C][H]) transposed to LDS [H][C], hi/lo ----
  #pragma unroll
  for (int w = 0; w < 3; ++w) {
    const float* Wg = (w == 0) ? Wk : (w == 1) ? Wq : Wv;
    unsigned short* dh = sWh + w * H * PX;
    unsigned short* dl = sWl + w * H * PX;
    for (int i = tid; i < C * H; i += NTHR) {
      int c = i >> 6, h = i & 63;
      float v = Wg[i];
      unsigned short hi = f2bf(v);
      dh[h * PX + c] = hi;
      dl[h * PX + c] = f2bf(v - bf2f(hi));
    }
  }
  __syncthreads();

  // ---- phase 1: K,Q,V = x @ W  (wave wv owns rows 16wv..16wv+15) ----
  v8f acc[12];
  #pragma unroll
  for (int t = 0; t < 12; ++t) acc[t] = v8f{};
  const int arow = 16 * wv + ml;
  for (int kk = 0; kk < C / 32; ++kk) {
    const int k0 = kk * 32;
    v16bf ah = loadA(sXh, arow, PX, k0, half);
    v16bf al = loadA(sXl, arow, PX, k0, half);
    #pragma unroll
    for (int w = 0; w < 3; ++w) {
      #pragma unroll
      for (int nt = 0; nt < 4; ++nt) {
        v16bf bh = loadB(sWh + w * H * PX, 16 * nt + ml, PX, k0, half);
        v16bf bl = loadB(sWl + w * H * PX, 16 * nt + ml, PX, k0, half);
        acc[w * 4 + nt] = wmma3(ah, al, bh, bl, acc[w * 4 + nt]);
      }
    }
  }
  __syncthreads();   // phase-1 LDS reads done; safe to alias

  // ---- scatter K, Q*(C^-1/2), V^T to LDS as hi/lo bf16 ----
  #pragma unroll
  for (int nt = 0; nt < 4; ++nt) {
    #pragma unroll
    for (int r = 0; r < 8; ++r) {
      const int mrow = 16 * wv + r + 8 * half;   // C/D frag: M = r + 8*(lane/16)
      const int hcol = 16 * nt + ml;             //            N = lane%16
      float kv = acc[0 * 4 + nt][r];
      unsigned short kh = f2bf(kv);
      sKh[mrow * PQK + hcol] = kh;
      sKl[mrow * PQK + hcol] = f2bf(kv - bf2f(kh));
      float qv = acc[1 * 4 + nt][r] * 0.0625f;   // fold C^-0.5 = 1/16 into Q
      unsigned short qh = f2bf(qv);
      sQh[mrow * PQK + hcol] = qh;
      sQl[mrow * PQK + hcol] = f2bf(qv - bf2f(qh));
      float vv = acc[2 * 4 + nt][r];
      unsigned short vh = f2bf(vv);
      sVth[hcol * PVT + mrow] = vh;              // V stored transposed [H][T]
      sVtl[hcol * PVT + mrow] = f2bf(vv - bf2f(vh));
    }
  }
  __syncthreads();

  // ---- phase 2a: wei = Q K^T (wave wv owns wei row-tile wv) ----
  {
    v8f wacc[6];
    #pragma unroll
    for (int t = 0; t < 6; ++t) wacc[t] = v8f{};
    for (int kk = 0; kk < H / 32; ++kk) {
      const int k0 = kk * 32;
      v16bf ah = loadA(sQh, arow, PQK, k0, half);
      v16bf al = loadA(sQl, arow, PQK, k0, half);
      #pragma unroll
      for (int n = 0; n < 6; ++n) {
        v16bf bh = loadB(sKh, 16 * n + ml, PQK, k0, half);
        v16bf bl = loadB(sKl, 16 * n + ml, PQK, k0, half);
        wacc[n] = wmma3(ah, al, bh, bl, wacc[n]);
      }
    }
    #pragma unroll
    for (int n = 0; n < 6; ++n)
      #pragma unroll
      for (int r = 0; r < 8; ++r)
        sWei[(16 * wv + r + 8 * half) * PW32 + 16 * n + ml] = wacc[n][r];
  }
  __syncthreads();

  // ---- causal softmax per row; emit hi/lo bf16 probs, zero above diagonal --
  if (tid < T) {
    const int t = tid;
    float* row = sWei + t * PW32;
    float mx = row[0];
    for (int s = 1; s <= t; ++s) mx = fmaxf(mx, row[s]);
    float sum = 0.f;
    for (int s = 0; s <= t; ++s) { float e = __expf(row[s] - mx); row[s] = e; sum += e; }
    float inv = 1.f / sum;
    for (int s = 0; s <= t; ++s) {
      float p = row[s] * inv;
      unsigned short ph = f2bf(p);
      sWbh[t * PWB + s] = ph;
      sWbl[t * PWB + s] = f2bf(p - bf2f(ph));
    }
    for (int s = t + 1; s < T; ++s) { sWbh[t * PWB + s] = 0; sWbl[t * PWB + s] = 0; }
  }
  __syncthreads();

  // ---- phase 2b: out = wei @ V ----
  {
    v8f oacc[4];
    #pragma unroll
    for (int t = 0; t < 4; ++t) oacc[t] = v8f{};
    for (int kk = 0; kk < T / 32; ++kk) {
      const int k0 = kk * 32;
      v16bf ah = loadA(sWbh, arow, PWB, k0, half);
      v16bf al = loadA(sWbl, arow, PWB, k0, half);
      #pragma unroll
      for (int n = 0; n < 4; ++n) {
        v16bf bh = loadB(sVth, 16 * n + ml, PVT, k0, half);
        v16bf bl = loadB(sVtl, 16 * n + ml, PVT, k0, half);
        oacc[n] = wmma3(ah, al, bh, bl, oacc[n]);
      }
    }
    float* ob = Out + (size_t)b * T * H;
    #pragma unroll
    for (int n = 0; n < 4; ++n)
      #pragma unroll
      for (int r = 0; r < 8; ++r)
        ob[(16 * wv + r + 8 * half) * H + 16 * n + ml] = oacc[n][r];
  }
}

extern "C" void kernel_launch(void* const* d_in, const int* in_sizes, int n_in,
                              void* d_out, int out_size, void* d_ws, size_t ws_size,
                              hipStream_t stream) {
  const float* x  = (const float*)d_in[0];
  const float* Wk = (const float*)d_in[1];
  const float* Wq = (const float*)d_in[2];
  const float* Wv = (const float*)d_in[3];
  float* out = (float*)d_out;
  const int B = in_sizes[0] / (T * C);   // 4096
  attn_head_wmma<<<dim3(B), dim3(NTHR), SMEM_BYTES, stream>>>(x, Wk, Wq, Wv, out);
}
